// Attention_61649960566834
// MI455X (gfx1250) — matrix operand order repaired and checked by reading, not compiled, not run on hardware
//
#include <hip/hip_runtime.h>
#include <stdint.h>

#define B_    2
#define S_    2048
#define DM    2048
#define NQ    32
#define NKV   8
#define HEADD 64
#define GROUP 4

typedef __attribute__((ext_vector_type(16))) __bf16 v16bf;
typedef __attribute__((ext_vector_type(8)))  float  v8f;

union Frag16 { v16bf v; uint32_t u[8]; };

__device__ __forceinline__ unsigned short f2bf(float x) {
  union { float f; uint32_t u; } a; a.f = x;
  uint32_t u = a.u;
  uint32_t r = (u + 0x7FFFu + ((u >> 16) & 1u)) >> 16;   // RNE
  return (unsigned short)r;
}
__device__ __forceinline__ uint32_t pack_bf2(float lo, float hi) {
  return (uint32_t)f2bf(lo) | ((uint32_t)f2bf(hi) << 16);
}

// Stage 8 consecutive elements to LDS as bf16 (16B store).
__device__ __forceinline__ void stage8(unsigned short* dst, const float* src) {
  float4 a = *(const float4*)src;
  float4 b = *(const float4*)(src + 4);
  uint4 p;
  p.x = pack_bf2(a.x, a.y);
  p.y = pack_bf2(a.z, a.w);
  p.z = pack_bf2(b.x, b.y);
  p.w = pack_bf2(b.z, b.w);
  *(uint4*)dst = p;
}
__device__ __forceinline__ void stage8(unsigned short* dst, const unsigned short* src) {
  *(uint4*)dst = *(const uint4*)src;
}

// CDNA5 async global->LDS copy (ASYNCcnt-tracked), 16 bytes per lane.
__device__ __forceinline__ void async_load_b128(uint32_t lds_byte_addr, const void* gaddr) {
  asm volatile("global_load_async_to_lds_b128 %0, %1, off"
               :: "v"(lds_byte_addr), "v"(gaddr) : "memory");
}
__device__ __forceinline__ void wait_asynccnt0() {
  asm volatile("s_wait_asynccnt 0" ::: "memory");
}

// ---------------------------------------------------------------------------
// Tiled GEMM: C[M,N] = A[M,K] * B[K,N] + bias, fp32 accum via bf16 WMMA.
// Block tile 64x64, BK=32, 256 threads = 8 waves, wave tile 16x32 (2 C frags).
// ---------------------------------------------------------------------------
template <typename AT>
__global__ __launch_bounds__(256) void gemm_bf16_kernel(
    const AT* __restrict__ A, const float* __restrict__ Bm,
    const float* __restrict__ bias, float* __restrict__ C,
    int M, int N, int K)
{
  __shared__ __align__(16) unsigned short As[64][32];   // [m][k]
  __shared__ __align__(16) unsigned short Bst[64][32];  // [n][k] (transposed)

  const int tid  = threadIdx.x;
  const int lane = tid & 31;
  const int w    = tid >> 5;
  const int wm   = w & 3;        // 4 waves along M
  const int wn   = w >> 2;       // 2 waves along N
  const int l16  = lane & 15;
  const int hl   = lane >> 4;    // lane half (wave32)
  const int bm   = blockIdx.y * 64;
  const int bn   = blockIdx.x * 64;

  // Per-thread staging coordinates
  const int ar = tid >> 2;            // A row 0..63
  const int ac = (tid & 3) * 8;       // A col 0,8,16,24
  const int br = tid >> 3;            // B k-row 0..31
  const int bc = (tid & 7) * 8;       // B n-col 0..56

  v8f c0 = {}; v8f c1 = {};

  for (int k0 = 0; k0 < K; k0 += 32) {
    __syncthreads();
    // Stage A tile 64x32: one 16B vector load + one 16B LDS store per thread
    stage8(&As[ar][ac], A + (size_t)(bm + ar) * K + (k0 + ac));
    // Stage B tile 32x64 transposed -> Bst[n][k]: vector load, scatter stores
    {
      const float* bp = Bm + (size_t)(k0 + br) * N + (bn + bc);
      float4 f0 = *(const float4*)bp;
      float4 f1 = *(const float4*)(bp + 4);
      Bst[bc + 0][br] = f2bf(f0.x);
      Bst[bc + 1][br] = f2bf(f0.y);
      Bst[bc + 2][br] = f2bf(f0.z);
      Bst[bc + 3][br] = f2bf(f0.w);
      Bst[bc + 4][br] = f2bf(f1.x);
      Bst[bc + 5][br] = f2bf(f1.y);
      Bst[bc + 6][br] = f2bf(f1.z);
      Bst[bc + 7][br] = f2bf(f1.w);
    }
    __syncthreads();

    // A fragment (16x32): VGPR v holds K={2v,2v+1}+16*half, M=lane%16
    Frag16 a;
#pragma unroll
    for (int v = 0; v < 8; ++v) {
      int kk = 2 * v + 16 * hl;
      a.u[v] = *(const uint32_t*)&As[wm * 16 + l16][kk];
    }
    // B fragments (32x16): VGPR v holds K={2v,2v+1}+16*half, N=lane%16
    Frag16 b0, b1;
#pragma unroll
    for (int v = 0; v < 8; ++v) {
      int kk = 2 * v + 16 * hl;
      b0.u[v] = *(const uint32_t*)&Bst[wn * 32 + l16][kk];
      b1.u[v] = *(const uint32_t*)&Bst[wn * 32 + 16 + l16][kk];
    }
    c0 = __builtin_amdgcn_wmma_f32_16x16x32_bf16(false, a.v, false, b0.v, (short)0, c0, false, false);
    c1 = __builtin_amdgcn_wmma_f32_16x16x32_bf16(false, a.v, false, b1.v, (short)0, c1, false, false);
  }

  // Epilogue: C/D frag row = r + 8*half, col = lane%16
#pragma unroll
  for (int r = 0; r < 8; ++r) {
    int mg = bm + wm * 16 + r + 8 * hl;
    int n0 = bn + wn * 32 + l16;
    int n1 = n0 + 16;
    C[(size_t)mg * N + n0] = c0[r] + bias[n0];
    C[(size_t)mg * N + n1] = c1[r] + bias[n1];
  }
}

// ---------------------------------------------------------------------------
// RoPE (interleaved pairs) + transpose (B,S,H,D) -> (B,H,S,D), fp32 -> bf16.
// One thread per (b,s,h,pair).
// ---------------------------------------------------------------------------
__global__ void rope_transpose_kernel(const float* __restrict__ src,
                                      const float* __restrict__ cosp,
                                      const float* __restrict__ sinp,
                                      unsigned short* __restrict__ dst,
                                      int H, int use_rope)
{
  long long t = (long long)blockIdx.x * blockDim.x + threadIdx.x;
  long long total = (long long)B_ * S_ * H * 32;
  if (t >= total) return;
  int j = t & 31;
  int h = (t >> 5) % H;
  int s = (int)((t / (32LL * H)) % S_);
  int b = (int)(t / (32LL * H * S_));

  size_t so = ((size_t)(b * S_ + s)) * ((size_t)H * HEADD) + (size_t)h * HEADD + 2 * j;
  float te = src[so], to = src[so + 1];
  float e, o;
  if (use_rope) {
    float c = cosp[s * 32 + j], si = sinp[s * 32 + j];
    e = te * c - to * si;
    o = te * si + to * c;
  } else { e = te; o = to; }
  size_t dofs = (((size_t)(b * H + h)) * S_ + s) * HEADD + 2 * j;
  *(uint32_t*)(dst + dofs) = pack_bf2(e, o);
}

// ---------------------------------------------------------------------------
// Flash attention (GQA): block = 128 threads = 4 waves, wave owns 16 query
// rows; block streams 32-key chunks of K/V through LDS (K via async-to-LDS).
// ---------------------------------------------------------------------------
__global__ __launch_bounds__(128) void attn_kernel(
    const unsigned short* __restrict__ qb,  // (B,NQ,S,64) bf16
    const unsigned short* __restrict__ kb,  // (B,NKV,S,64) bf16 (roped)
    const unsigned short* __restrict__ vb,  // (B,NKV,S,64) bf16
    unsigned short* __restrict__ ob)        // (B,S,NQ*64) bf16
{
  __shared__ __align__(16) unsigned short Ks[32][64];   // [key][dim]
  __shared__ __align__(16) unsigned short Vst[64][32];  // [dim][key]
  __shared__ __align__(16) float          Pw[4][16][32];
  __shared__ __align__(16) unsigned short Pb[4][16][32];
  __shared__ __align__(16) float          Arow[4][16];

  const int tid  = threadIdx.x;
  const int lane = tid & 31;
  const int w    = tid >> 5;
  const int l16  = lane & 15;
  const int hl   = lane >> 4;

  const int bid = blockIdx.x;
  const int qt  = bid & 31;                 // S/64 = 32 query tiles
  const int h   = (bid >> 5) % NQ;
  const int b   = bid / (32 * NQ);
  const int kv  = h / GROUP;
  const int row0 = qt * 64 + w * 16;        // wave's first query row

  const unsigned short* qbase = qb + (((size_t)(b * NQ + h)) * S_ + row0) * HEADD;
  const unsigned short* kbase = kb + (((size_t)(b * NKV + kv)) * S_) * HEADD;
  const unsigned short* vbase = vb + (((size_t)(b * NKV + kv)) * S_) * HEADD;

  const uint32_t ldsK = (uint32_t)(uintptr_t)&Ks[0][0];

  // Load Q fragments once: 16x64 = two 16x32 A-frags
  Frag16 qa0, qa1;
#pragma unroll
  for (int v = 0; v < 8; ++v) {
    int kk = 2 * v + 16 * hl;
    qa0.u[v] = *(const uint32_t*)(qbase + (size_t)l16 * HEADD + kk);
    qa1.u[v] = *(const uint32_t*)(qbase + (size_t)l16 * HEADD + 32 + kk);
  }

  v8f o0 = {}, o1 = {}, o2 = {}, o3 = {};
  float mrow = -1e30f, lrow = 0.0f;
  const float scale = 0.125f;               // 1/sqrt(64)

  for (int t = 0; t < S_ / 32; ++t) {
    __syncthreads();
    // Stage K chunk (32x64 bf16 = 4KB) via CDNA5 async global->LDS copies:
    // 128 lanes x 16B x 2 issues, no VGPR round-trip, ASYNCcnt-tracked.
    {
      const char* gk = (const char*)(kbase + (size_t)t * 32 * HEADD);
      async_load_b128(ldsK + tid * 16,        gk + tid * 16);
      async_load_b128(ldsK + 2048 + tid * 16, gk + 2048 + tid * 16);
    }
    // Stage V chunk transposed: Vst[dim][key] (dword global reads)
    {
      const uint32_t* vsrc = (const uint32_t*)(vbase + (size_t)t * 32 * HEADD);
#pragma unroll
      for (int i = 0; i < 8; ++i) {
        int e = tid + i * 128;               // dword index over 32x64/2
        int key = e >> 5, d = (e & 31) * 2;
        uint32_t val = vsrc[e];
        Vst[d][key]     = (unsigned short)(val & 0xFFFFu);
        Vst[d + 1][key] = (unsigned short)(val >> 16);
      }
    }
    wait_asynccnt0();
    __syncthreads();

    // Scores: S(16x32) = Q(16x64) x K^T(64x32), 4 WMMAs
    v8f s0 = {}, s1 = {};
    {
      Frag16 bk;
#pragma unroll
      for (int v = 0; v < 8; ++v) { int kk = 2 * v + 16 * hl; bk.u[v] = *(const uint32_t*)&Ks[l16][kk]; }
      s0 = __builtin_amdgcn_wmma_f32_16x16x32_bf16(false, qa0.v, false, bk.v, (short)0, s0, false, false);
#pragma unroll
      for (int v = 0; v < 8; ++v) { int kk = 2 * v + 16 * hl; bk.u[v] = *(const uint32_t*)&Ks[l16][32 + kk]; }
      s0 = __builtin_amdgcn_wmma_f32_16x16x32_bf16(false, qa1.v, false, bk.v, (short)0, s0, false, false);
#pragma unroll
      for (int v = 0; v < 8; ++v) { int kk = 2 * v + 16 * hl; bk.u[v] = *(const uint32_t*)&Ks[16 + l16][kk]; }
      s1 = __builtin_amdgcn_wmma_f32_16x16x32_bf16(false, qa0.v, false, bk.v, (short)0, s1, false, false);
#pragma unroll
      for (int v = 0; v < 8; ++v) { int kk = 2 * v + 16 * hl; bk.u[v] = *(const uint32_t*)&Ks[16 + l16][32 + kk]; }
      s1 = __builtin_amdgcn_wmma_f32_16x16x32_bf16(false, qa1.v, false, bk.v, (short)0, s1, false, false);
    }

    // Spill scaled scores to LDS (also serves as C-frag -> A-frag transpose)
#pragma unroll
    for (int r = 0; r < 8; ++r) {
      int row = r + 8 * hl;
      Pw[w][row][l16]      = s0[r] * scale;
      Pw[w][row][16 + l16] = s1[r] * scale;
    }
    __syncthreads();

    // Online softmax: lanes 0-15 each own one query row
    if (lane < 16) {
      float mx = -1e30f;
#pragma unroll
      for (int c = 0; c < 32; ++c) mx = fmaxf(mx, Pw[w][lane][c]);
      float mnew  = fmaxf(mrow, mx);
      float alpha = __expf(mrow - mnew);
      float sum = 0.0f;
#pragma unroll
      for (int c = 0; c < 32; c += 2) {
        float p0 = __expf(Pw[w][lane][c] - mnew);
        float p1 = __expf(Pw[w][lane][c + 1] - mnew);
        sum += p0 + p1;
        *(uint32_t*)&Pb[w][lane][c] = pack_bf2(p0, p1);
      }
      lrow = lrow * alpha + sum;
      mrow = mnew;
      Arow[w][lane] = alpha;
    }
    __syncthreads();

    // Rescale O accumulators by per-row alpha
    float al[8];
#pragma unroll
    for (int r = 0; r < 8; ++r) al[r] = Arow[w][r + 8 * hl];
#pragma unroll
    for (int r = 0; r < 8; ++r) { o0[r] *= al[r]; o1[r] *= al[r]; o2[r] *= al[r]; o3[r] *= al[r]; }

    // O += P(16x32) x V(32x64): 4 WMMAs
    Frag16 pa;
#pragma unroll
    for (int v = 0; v < 8; ++v) {
      int kk = 2 * v + 16 * hl;
      pa.u[v] = *(const uint32_t*)&Pb[w][l16][kk];
    }
    Frag16 bv;
#pragma unroll
    for (int v = 0; v < 8; ++v) { int kk = 2 * v + 16 * hl; bv.u[v] = *(const uint32_t*)&Vst[l16][kk]; }
    o0 = __builtin_amdgcn_wmma_f32_16x16x32_bf16(false, pa.v, false, bv.v, (short)0, o0, false, false);
#pragma unroll
    for (int v = 0; v < 8; ++v) { int kk = 2 * v + 16 * hl; bv.u[v] = *(const uint32_t*)&Vst[16 + l16][kk]; }
    o1 = __builtin_amdgcn_wmma_f32_16x16x32_bf16(false, pa.v, false, bv.v, (short)0, o1, false, false);
#pragma unroll
    for (int v = 0; v < 8; ++v) { int kk = 2 * v + 16 * hl; bv.u[v] = *(const uint32_t*)&Vst[32 + l16][kk]; }
    o2 = __builtin_amdgcn_wmma_f32_16x16x32_bf16(false, pa.v, false, bv.v, (short)0, o2, false, false);
#pragma unroll
    for (int v = 0; v < 8; ++v) { int kk = 2 * v + 16 * hl; bv.u[v] = *(const uint32_t*)&Vst[48 + l16][kk]; }
    o3 = __builtin_amdgcn_wmma_f32_16x16x32_bf16(false, pa.v, false, bv.v, (short)0, o3, false, false);
  }

  __syncthreads();
  if (lane < 16) Arow[w][lane] = 1.0f / lrow;
  __syncthreads();

  float il[8];
#pragma unroll
  for (int r = 0; r < 8; ++r) il[r] = Arow[w][r + 8 * hl];

#pragma unroll
  for (int r = 0; r < 8; ++r) {
    int srow = row0 + r + 8 * hl;
    size_t base = ((size_t)(b * S_ + srow)) * DM + (size_t)h * HEADD + l16;
    ob[base + 0]  = f2bf(o0[r] * il[r]);
    ob[base + 16] = f2bf(o1[r] * il[r]);
    ob[base + 32] = f2bf(o2[r] * il[r]);
    ob[base + 48] = f2bf(o3[r] * il[r]);
  }
}

// ---------------------------------------------------------------------------
extern "C" void kernel_launch(void* const* d_in, const int* in_sizes, int n_in,
                              void* d_out, int out_size, void* d_ws, size_t ws_size,
                              hipStream_t stream)
{
  const float* x    = (const float*)d_in[0];
  const float* cosp = (const float*)d_in[1];
  const float* sinp = (const float*)d_in[2];
  const float* wq   = (const float*)d_in[3];
  const float* bq   = (const float*)d_in[4];
  const float* wk   = (const float*)d_in[5];
  const float* bk   = (const float*)d_in[6];
  const float* wv   = (const float*)d_in[7];
  const float* bv   = (const float*)d_in[8];
  const float* wo   = (const float*)d_in[9];
  const float* bo   = (const float*)d_in[10];
  float* out = (float*)d_out;

  const size_t M = (size_t)B_ * S_;     // 4096

  char* ws = (char*)d_ws;
  size_t off = 0;
  auto alloc = [&](size_t bytes) { void* p = ws + off; off += (bytes + 255) & ~(size_t)255; return p; };

  float* qf = (float*)alloc(M * DM * 4);
  float* kf = (float*)alloc(M * (NKV * HEADD) * 4);
  float* vf = (float*)alloc(M * (NKV * HEADD) * 4);
  unsigned short* qbf = (unsigned short*)alloc(M * DM * 2);
  unsigned short* kbf = (unsigned short*)alloc(M * (NKV * HEADD) * 2);
  unsigned short* vbf = (unsigned short*)alloc(M * (NKV * HEADD) * 2);
  unsigned short* obf = (unsigned short*)alloc(M * DM * 2);

  dim3 blk(256);

  // QKV projections (+bias) in bf16 WMMA, fp32 out
  gemm_bf16_kernel<float><<<dim3(DM / 64, M / 64), blk, 0, stream>>>(x, wq, bq, qf, (int)M, DM, DM);
  gemm_bf16_kernel<float><<<dim3((NKV * HEADD) / 64, M / 64), blk, 0, stream>>>(x, wk, bk, kf, (int)M, NKV * HEADD, DM);
  gemm_bf16_kernel<float><<<dim3((NKV * HEADD) / 64, M / 64), blk, 0, stream>>>(x, wv, bv, vf, (int)M, NKV * HEADD, DM);

  // RoPE + transpose to head-major bf16
  {
    long long tq = (long long)B_ * S_ * NQ * 32;
    long long tk = (long long)B_ * S_ * NKV * 32;
    rope_transpose_kernel<<<(unsigned)((tq + 255) / 256), 256, 0, stream>>>(qf, cosp, sinp, qbf, NQ, 1);
    rope_transpose_kernel<<<(unsigned)((tk + 255) / 256), 256, 0, stream>>>(kf, cosp, sinp, kbf, NKV, 1);
    rope_transpose_kernel<<<(unsigned)((tk + 255) / 256), 256, 0, stream>>>(vf, cosp, sinp, vbf, NKV, 0);
  }

  // Flash attention: one block per (b, head, 64-row query tile)
  attn_kernel<<<B_ * NQ * (S_ / 64), 128, 0, stream>>>(qbf, kbf, vbf, obf);

  // Output projection (+bo), bf16 A path
  gemm_bf16_kernel<unsigned short><<<dim3(DM / 64, M / 64), blk, 0, stream>>>(obf, wo, bo, out, (int)M, DM, DM);
}
